// SelfAttention_33337536151970
// MI455X (gfx1250) — matrix operand order repaired and checked by reading, not compiled, not run on hardware
//
#include <hip/hip_runtime.h>

typedef __attribute__((ext_vector_type(16))) _Float16 v16h;
typedef __attribute__((ext_vector_type(8)))  _Float16 v8h;
typedef __attribute__((ext_vector_type(8)))  float    v8f;

#define E_DIM 1024
#define S_DIM 2048
#define B_DIM 4
#define H_DIM 16
#define D_DIM 64
#define BSROWS (B_DIM * S_DIM)   // 8192

#define BM 128
#define BN 64
#define BK 32

static __device__ inline v16h cat16(v8h lo, v8h hi) {
  return __builtin_shufflevector(lo, hi, 0,1,2,3,4,5,6,7,8,9,10,11,12,13,14,15);
}

// NaN-free max (avoids v_max_num canonicalize pairs)
static __device__ inline float maxf(float a, float b) { return a > b ? a : b; }

// raw v_exp_f32 (no denorm fixup; -1e30 underflows to 0, which is what we want)
static __device__ inline float fast_exp2(float x) {
#if __has_builtin(__builtin_amdgcn_exp2f)
  return __builtin_amdgcn_exp2f(x);
#else
  return exp2f(x);
#endif
}

static __device__ inline float fast_rcp(float x) {
#if __has_builtin(__builtin_amdgcn_rcpf)
  return __builtin_amdgcn_rcpf(x);
#else
  return 1.0f / x;
#endif
}

#if __has_builtin(__builtin_amdgcn_update_dpp)
template <int CTRL>
static __device__ inline float dppmaxstep(float x) {
  int yi = __builtin_amdgcn_update_dpp(0, __builtin_bit_cast(int, x),
                                       CTRL, 0xF, 0xF, true);
  float y = __builtin_bit_cast(float, yi);
  return x > y ? x : y;
}
// butterfly max over each 16-lane DPP row (our half-wave groups)
static __device__ inline float rmax16(float x) {
  x = dppmaxstep<0xB1>(x);    // quad_perm(1,0,3,2)  : xor 1
  x = dppmaxstep<0x4E>(x);    // quad_perm(2,3,0,1)  : xor 2
  x = dppmaxstep<0x141>(x);   // row_half_mirror     : xor 4 (post-reduce)
  x = dppmaxstep<0x140>(x);   // row_mirror          : xor 8 (post-reduce)
  return x;
}
#else
static __device__ inline float rmax16(float x) {
#pragma unroll
  for (int off = 1; off < 16; off <<= 1) {
    float y = __shfl_xor(x, off, 32);
    x = maxf(x, y);
  }
  return x;
}
#endif

// ---------------- fp32 -> fp16 conversion ----------------
__global__ void cvt_f32_f16(const float* __restrict__ in,
                            _Float16* __restrict__ out, int n) {
  int i = blockIdx.x * blockDim.x + threadIdx.x;
  int stride = gridDim.x * blockDim.x;
  for (; i < n; i += stride) out[i] = (_Float16)in[i];
}

// ---------------- tiled WMMA GEMM ----------------
// C[M=8192, N=1024] = A[M,1024](f16) * W[1024,N](f16) + bias
// MODE 0: scale + scatter to [B*H, S, D] f16   (Q, K)
// MODE 2: scale + scatter to [B*H, D, S] f16   (V transposed)
// MODE 1: plain fp32 row-major output          (final projection)
template <int MODE>
__global__ __launch_bounds__(256)
void gemm_wmma(const _Float16* __restrict__ A,
               const _Float16* __restrict__ W,
               const float* __restrict__ bias,
               void* __restrict__ outv,
               float scale) {
  __shared__ _Float16 As [BM][BK + 8];   // row pitch 40 f16 = 80B (16B aligned)
  __shared__ _Float16 Bst[BN][BK + 8];   // B tile stored transposed [n][k]

  const int tid  = threadIdx.x;
  const int m0   = blockIdx.y * BM;
  const int n0   = blockIdx.x * BN;
  const int wave = __builtin_amdgcn_readfirstlane(tid >> 5);
  const int lane = tid & 31;
  const int ln   = lane & 15;
  const int hgrp = lane >> 4;
  const int wr   = wave >> 1;   // 0..3 -> 32-row stripe
  const int wc   = wave & 1;    // 0..1 -> 32-col stripe

  v8f acc[2][2] = {};

  for (int k0 = 0; k0 < E_DIM; k0 += BK) {
    // stage A tile: 128x32, 16 f16 per thread, b128 in / b128 out
    {
      int r = tid >> 1, seg = (tid & 1) * 16;
      const _Float16* src = A + (size_t)(m0 + r) * E_DIM + k0 + seg;
      *(v8h*)&As[r][seg]     = *(const v8h*)src;
      *(v8h*)&As[r][seg + 8] = *(const v8h*)(src + 8);
    }
    // stage B tile transposed: b128 global load, 8x b16 LDS scatter
    {
      int kr = tid >> 3, nseg = (tid & 7) * 8;
      const _Float16* src = W + (size_t)(k0 + kr) * E_DIM + n0 + nseg;
      v8h w = *(const v8h*)src;
#pragma unroll
      for (int i = 0; i < 8; ++i) Bst[nseg + i][kr] = w[i];
      if (k0 + BK < E_DIM) __builtin_prefetch(src + (size_t)BK * E_DIM, 0, 1);
    }
    __syncthreads();

    v16h a[2], b[2];
#pragma unroll
    for (int i = 0; i < 2; ++i) {      // A frag: lane-half -> K 0-7/16-23 vs 8-15/24-31
      int row = wr * 32 + i * 16 + ln;
      int kb  = hgrp * 8;
      v8h lo = *(const v8h*)&As[row][kb];
      v8h hi = *(const v8h*)&As[row][kb + 16];
      a[i] = cat16(lo, hi);
    }
#pragma unroll
    for (int j = 0; j < 2; ++j) {      // B frag: lane-half -> K 0-7/16-23 vs 8-15/24-31
      int col = wc * 32 + j * 16 + ln;
      int kh  = hgrp * 8;
      v8h lo = *(const v8h*)&Bst[col][kh];
      v8h hi = *(const v8h*)&Bst[col][kh + 16];
      b[j] = cat16(lo, hi);
    }
#pragma unroll
    for (int i = 0; i < 2; ++i)
#pragma unroll
      for (int j = 0; j < 2; ++j)
        acc[i][j] = __builtin_amdgcn_wmma_f32_16x16x32_f16(
            false, a[i], false, b[j], (short)0, acc[i][j], false, false);
    __syncthreads();
  }

  // epilogue
#pragma unroll
  for (int i = 0; i < 2; ++i) {
#pragma unroll
    for (int j = 0; j < 2; ++j) {
      int gn = n0 + wc * 32 + j * 16 + ln;
      float bv = bias[gn];
#pragma unroll
      for (int v = 0; v < 8; ++v) {
        int gm = m0 + wr * 32 + i * 16 + hgrp * 8 + v;
        float val = acc[i][j][v] + bv;
        if (MODE == 1) {
          ((float*)outv)[(size_t)gm * E_DIM + gn] = val;
        } else {
          val *= scale;
          int bb = gm >> 11, s = gm & (S_DIM - 1);
          int h = gn >> 6, d = gn & (D_DIM - 1);
          _Float16* o = (_Float16*)outv;
          if (MODE == 0)
            o[(((size_t)bb * H_DIM + h) * S_DIM + s) * D_DIM + d] = (_Float16)val;
          else
            o[(((size_t)bb * H_DIM + h) * D_DIM + d) * S_DIM + s] = (_Float16)val;
        }
      }
    }
  }
}

// ---------------- flash attention (causal), 64-wide KV blocks ----------------
// Q [B*H, S, D] f16 (pre-scaled by log2e/sqrt(D)), K [B*H, S, D], Vt [B*H, D, S]
// O [B, S, E] f16. Softmax in base-2 domain (raw v_exp_f32).
__global__ __launch_bounds__(256)
void attn_wmma(const _Float16* __restrict__ Q,
               const _Float16* __restrict__ K,
               const _Float16* __restrict__ Vt,
               _Float16* __restrict__ O) {
  __shared__ _Float16 P[8][16][72];   // per-wave 16x64 P tile, pitch 72 f16 (144B)

  const int tid  = threadIdx.x;
  const int wave = __builtin_amdgcn_readfirstlane(tid >> 5);
  const int lane = tid & 31;
  const int ln   = lane & 15;
  const int hgrp = lane >> 4;
  const int bh   = blockIdx.y;
  const int q0   = __builtin_amdgcn_readfirstlane((int)blockIdx.x * 128 + wave * 16);

  const _Float16* Qb = Q  + (size_t)bh * S_DIM * D_DIM;
  const _Float16* Kb = K  + (size_t)bh * S_DIM * D_DIM;
  const _Float16* Vb = Vt + (size_t)bh * D_DIM * S_DIM;

  // Q fragments for the wave's 16 rows (two K=32 steps over D=64)
  v16h aq[2];
  {
    int row = q0 + ln;
#pragma unroll
    for (int t = 0; t < 2; ++t) {
      int kb = t * 32 + hgrp * 8;
      v8h lo = *(const v8h*)(Qb + (size_t)row * D_DIM + kb);
      v8h hi = *(const v8h*)(Qb + (size_t)row * D_DIM + kb + 16);
      aq[t] = cat16(lo, hi);
    }
  }

  // all-ones B fragment: row-sum of P via WMMA
  v16h onesb;
#pragma unroll
  for (int i = 0; i < 16; ++i) onesb[i] = (_Float16)1.0f;

  float m_i[8], l_i[8];
#pragma unroll
  for (int v = 0; v < 8; ++v) { m_i[v] = -1e30f; l_i[v] = 0.0f; }
  v8f o[4] = {};

  // causal: KV blocks of 64; last block always straddles the diagonal
  const int nblk = (q0 >> 6) + 1;
  for (int blk = 0; blk < nblk; ++blk) {
    const int kv0 = blk * 64;

    // scores S = Q * K^T  -> four 16x16 f32 tiles (16 x 64)
    v8f s[4] = {};
#pragma unroll
    for (int j = 0; j < 4; ++j) {
      int kvpos = kv0 + j * 16 + ln;
#pragma unroll
      for (int t = 0; t < 2; ++t) {
        int kh = t * 32 + hgrp * 8;
        const _Float16* src = Kb + (size_t)kvpos * D_DIM + kh;
        v16h kb = cat16(*(const v8h*)src, *(const v8h*)(src + 16));
        s[j] = __builtin_amdgcn_wmma_f32_16x16x32_f16(
            false, aq[t], false, kb, (short)0, s[j], false, false);
      }
    }

    // only the last block is partially masked (scalar condition)
    const bool edge = (blk == nblk - 1);
#pragma unroll
    for (int v = 0; v < 8; ++v) {
      int row = q0 + hgrp * 8 + v;
      if (edge) {
#pragma unroll
        for (int j = 0; j < 4; ++j)
          if (kv0 + j * 16 + ln > row) s[j][v] = -1e30f;
      }
      float mx = maxf(maxf(s[0][v], s[1][v]), maxf(s[2][v], s[3][v]));
      mx = rmax16(mx);
      float mnew  = maxf(m_i[v], mx);
      float alpha = fast_exp2(m_i[v] - mnew);
      l_i[v] *= alpha;
      m_i[v] = mnew;
#pragma unroll
      for (int n = 0; n < 4; ++n) o[n][v] *= alpha;
#pragma unroll
      for (int j = 0; j < 4; ++j)
        P[wave][hgrp * 8 + v][j * 16 + ln] = (_Float16)fast_exp2(s[j][v] - mnew);
    }

    // reload P as two A fragments (C->A relayout through per-wave LDS)
    v16h ap[2];
#pragma unroll
    for (int t = 0; t < 2; ++t) {
      int kb = t * 32 + hgrp * 8;
      v8h lo = *(const v8h*)&P[wave][ln][kb];
      v8h hi = *(const v8h*)&P[wave][ln][kb + 16];
      ap[t] = cat16(lo, hi);
    }

    // row-sum of P via WMMA against ones (replaces cross-lane sum reduce)
    {
      v8f rs = {};
#pragma unroll
      for (int t = 0; t < 2; ++t)
        rs = __builtin_amdgcn_wmma_f32_16x16x32_f16(
            false, ap[t], false, onesb, (short)0, rs, false, false);
#pragma unroll
      for (int v = 0; v < 8; ++v) l_i[v] += rs[v];
    }

    // O += P * V  (V transposed: two contiguous 8-f16 loads per lane)
#pragma unroll
    for (int n = 0; n < 4; ++n) {
      int dcol = n * 16 + ln;
#pragma unroll
      for (int t = 0; t < 2; ++t) {
        int kh = t * 32 + hgrp * 8;
        const _Float16* src = Vb + (size_t)dcol * S_DIM + kv0 + kh;
        v16h vb = cat16(*(const v8h*)src, *(const v8h*)(src + 16));
        o[n] = __builtin_amdgcn_wmma_f32_16x16x32_f16(
            false, ap[t], false, vb, (short)0, o[n], false, false);
      }
    }
  }

  // write O / l  -> [B, S, E] f16
  int bb = bh >> 4, h = bh & (H_DIM - 1);
#pragma unroll
  for (int v = 0; v < 8; ++v) {
    float inv = fast_rcp(l_i[v]);
    int row = q0 + hgrp * 8 + v;
#pragma unroll
    for (int n = 0; n < 4; ++n) {
      int d = n * 16 + ln;
      O[((size_t)bb * S_DIM + row) * E_DIM + h * D_DIM + d] =
          (_Float16)(o[n][v] * inv);
    }
  }
}

// ---------------- launch ----------------
extern "C" void kernel_launch(void* const* d_in, const int* in_sizes, int n_in,
                              void* d_out, int out_size, void* d_ws, size_t ws_size,
                              hipStream_t stream) {
  const float* x  = (const float*)d_in[0];
  const float* Wq = (const float*)d_in[1];
  const float* bq = (const float*)d_in[2];
  const float* Wk = (const float*)d_in[3];
  const float* bk = (const float*)d_in[4];
  const float* Wv = (const float*)d_in[5];
  const float* bv = (const float*)d_in[6];
  const float* Wo = (const float*)d_in[7];
  const float* bo = (const float*)d_in[8];
  float* out = (float*)d_out;

  char* ws = (char*)d_ws;
  const size_t xhBytes = (size_t)BSROWS * E_DIM * 2;          // 16 MiB
  const size_t wBytes  = (size_t)E_DIM * E_DIM * 2;           //  2 MiB
  _Float16* xh  = (_Float16*)(ws);
  _Float16* wqh = (_Float16*)(ws + xhBytes);
  _Float16* wkh = (_Float16*)(ws + xhBytes + wBytes);
  _Float16* wvh = (_Float16*)(ws + xhBytes + 2 * wBytes);
  _Float16* woh = (_Float16*)(ws + xhBytes + 3 * wBytes);
  _Float16* Qh  = (_Float16*)(ws + xhBytes + 4 * wBytes);
  _Float16* Kh  = (_Float16*)(ws + 2 * xhBytes + 4 * wBytes);
  _Float16* Vh  = (_Float16*)(ws + 3 * xhBytes + 4 * wBytes);
  _Float16* Oh  = (_Float16*)(ws + 4 * xhBytes + 4 * wBytes);

  // 1) convert inputs to f16
  cvt_f32_f16<<<4096, 256, 0, stream>>>(x,  xh,  BSROWS * E_DIM);
  cvt_f32_f16<<<1024, 256, 0, stream>>>(Wq, wqh, E_DIM * E_DIM);
  cvt_f32_f16<<<1024, 256, 0, stream>>>(Wk, wkh, E_DIM * E_DIM);
  cvt_f32_f16<<<1024, 256, 0, stream>>>(Wv, wvh, E_DIM * E_DIM);
  cvt_f32_f16<<<1024, 256, 0, stream>>>(Wo, woh, E_DIM * E_DIM);

  // 2) QKV projections.
  //    Q pre-scaled by log2e/sqrt(D) so softmax runs in exp2 domain.
  dim3 gGemm(E_DIM / BN, BSROWS / BM);   // (16, 64)
  gemm_wmma<0><<<gGemm, 256, 0, stream>>>(xh, wqh, bq, Qh, 0.125f * 1.44269504f);
  gemm_wmma<0><<<gGemm, 256, 0, stream>>>(xh, wkh, bk, Kh, 1.0f);
  gemm_wmma<2><<<gGemm, 256, 0, stream>>>(xh, wvh, bv, Vh, 1.0f);

  // 3) causal flash attention (64-wide KV blocks)
  dim3 gAttn(S_DIM / 128, B_DIM * H_DIM);  // (16, 64)
  attn_wmma<<<gAttn, 256, 0, stream>>>(Qh, Kh, Vh, Oh);

  // 4) output projection -> fp32
  gemm_wmma<1><<<gGemm, 256, 0, stream>>>(Oh, woh, bo, out, 1.0f);
}
